// MHSA_Intra_9380208575355
// MI455X (gfx1250) — matrix operand order repaired and checked by reading, not compiled, not run on hardware
//
#include <hip/hip_runtime.h>
#include <hip/hip_bf16.h>

#define Bx 8
#define Cx 1024
#define Tx 1024
#define Hx 16
#define DHx 64
#define SCALEx 0.125f
#define EPSx 1e-5f

typedef unsigned short u16;
typedef __attribute__((ext_vector_type(16))) _Float16 v16h;
typedef __attribute__((ext_vector_type(8)))  float    v8f;
typedef __attribute__((ext_vector_type(4)))  unsigned u32x4;

union Frag { v16h h; u32x4 q[2]; unsigned u[8]; };

__device__ __forceinline__ u16 f2h(float x) {
    _Float16 hv = (_Float16)x;
    union { _Float16 f; u16 u; } c; c.f = hv; return c.u;
}

// ------------------------------------------------- fp32 -> fp16 (contiguous, for weights)
__global__ void cvt_f32_f16(const float* __restrict__ in, u16* __restrict__ out, int n) {
    int i = blockIdx.x * 256 + threadIdx.x;
    if (i < n) out[i] = f2h(in[i]);
}

// ------------------------------------------------- x (B,C,T) fp32 -> XT (B,T,C) fp16, LDS-tiled
__global__ __launch_bounds__(256) void cvt_transpose(const float* __restrict__ in,
                                                     u16* __restrict__ out) {
    __shared__ u16 tile[32][33];
    const int t0 = blockIdx.x * 32, c0 = blockIdx.y * 32;
    const size_t b = blockIdx.z;
    const float* inb = in + b * (size_t)Cx * Tx;
    u16* outb = out + b * (size_t)Tx * Cx;
#pragma unroll
    for (int r = threadIdx.y; r < 32; r += 8)
        tile[r][threadIdx.x] = f2h(inb[(size_t)(c0 + r) * Tx + t0 + threadIdx.x]);
    __syncthreads();
#pragma unroll
    for (int r = threadIdx.y; r < 32; r += 8)
        outb[(size_t)(t0 + r) * Cx + c0 + threadIdx.x] = tile[threadIdx.x][r];
}

// ------------------------------------------------- GEMM: out(o,n) = W(o,k) * X^T(n,k) + bias
// Wave computes 32(M) x 64(N): 2 A-frags x 4 B-frags -> 8 WMMAs / k-step (1.5 b128/WMMA).
// OUT_MODE: 0 = fp32 c-major (B,C,T); 1 = fp16 c-major (B,C,T); 2 = fp16 n-major (B,T,C)
template <int OUT_MODE>
__global__ __launch_bounds__(128) void gemm_proj(
    const u16* __restrict__ W,
    const u16* __restrict__ X,     // (B, Tx, Cx) fp16 bits, n-major
    const float* __restrict__ bias,
    float scale,
    u16* __restrict__ outH, float* __restrict__ outF)
{
    const int lane = threadIdx.x;
    const int m    = lane & 15;
    const int hl   = lane >> 4;
    const int m0   = blockIdx.y * 128 + threadIdx.y * 32;
    const int n0   = blockIdx.x * 64;
    const size_t b = blockIdx.z;
    const u16* Xb  = X + b * (size_t)Tx * Cx;

    v8f zero = {};
    v8f acc[2][4] = { { zero, zero, zero, zero }, { zero, zero, zero, zero } };

    const u16* arow[2];
    arow[0] = W + (size_t)(m0 + m) * Cx;
    arow[1] = W + (size_t)(m0 + 16 + m) * Cx;
    const u16* brow[4];
#pragma unroll
    for (int j = 0; j < 4; ++j) brow[j] = Xb + (size_t)(n0 + j * 16 + m) * Cx;

    for (int k0 = 0; k0 < Cx; k0 += 32) {
        Frag a[2];   // A: VGPR0-7 = halfs [hl*16, hl*16+16) of row m
#pragma unroll
        for (int i = 0; i < 2; ++i) {
            a[i].q[0] = *(const u32x4*)(arow[i] + k0 + (hl << 4));
            a[i].q[1] = *(const u32x4*)(arow[i] + k0 + 8 + (hl << 4));
        }
        __builtin_prefetch((const void*)(arow[0] + k0 + 32), 0, 0);
        __builtin_prefetch((const void*)(brow[0] + k0 + 32), 0, 0);
#pragma unroll
        for (int j = 0; j < 4; ++j) {
            Frag bf;  // B: VGPR0-7 = halfs [hl*16, hl*16+16) of column n
            const u16* p = brow[j] + k0 + (hl << 4);
            bf.q[0] = *(const u32x4*)(p);
            bf.q[1] = *(const u32x4*)(p + 8);
#pragma unroll
            for (int i = 0; i < 2; ++i)
                acc[i][j] = __builtin_amdgcn_wmma_f32_16x16x32_f16(
                    false, a[i].h, false, bf.h, (short)0, acc[i][j], false, false);
        }
    }

#pragma unroll
    for (int i = 0; i < 2; ++i)
#pragma unroll
        for (int v = 0; v < 8; ++v) {
            int o = m0 + i * 16 + v + (hl << 3);
            float bia = bias[o];
#pragma unroll
            for (int j = 0; j < 4; ++j) {
                float val = (acc[i][j][v] + bia) * scale;
                int n = n0 + j * 16 + m;
                if (OUT_MODE == 0)      outF[(b * Cx + o) * (size_t)Tx + n] = val;
                else if (OUT_MODE == 1) outH[(b * Cx + o) * (size_t)Tx + n] = f2h(val);
                else                    outH[(b * Tx + n) * (size_t)Cx + o] = f2h(val);
            }
        }
}

// ------------------------------------------------- flash attention (64 keys / iteration)
// QT/KT: (B,T,C) t-major; V: (B,C,T) c-major; O written t-major (B,T,C).
// Row sums computed by WMMA against an all-ones fragment (no sum butterflies).
__global__ __launch_bounds__(32) void flash_attn(
    const u16* __restrict__ QT, const u16* __restrict__ KT,
    const u16* __restrict__ V,  u16* __restrict__ OT)
{
    const int lane = threadIdx.x;
    const int m    = lane & 15;
    const int hl   = lane >> 4;
    const int t0   = blockIdx.x * 16;
    const int h    = blockIdx.y;
    const size_t b = blockIdx.z;

    const u16* qt = QT + b * (size_t)Tx * Cx + h * DHx;   // (t, d) rows, stride Cx
    const u16* kt = KT + b * (size_t)Tx * Cx + h * DHx;   // (s, d) rows, stride Cx
    const u16* vb = V  + (b * Cx + (size_t)h * DHx) * Tx; // (d, s) rows, stride Tx
    u16*       ot = OT + b * (size_t)Tx * Cx + h * DHx;   // (t, d) rows, stride Cx

    // Q^T A-fragments (row t0+m, contiguous d)
    Frag aq[2];
    {
        const u16* qr = qt + (size_t)(t0 + m) * Cx;
#pragma unroll
        for (int f = 0; f < 2; ++f) {
            aq[f].q[0] = *(const u32x4*)(qr + f * 32 + (hl << 4));
            aq[f].q[1] = *(const u32x4*)(qr + f * 32 + 8 + (hl << 4));
        }
    }

    Frag ones;   // fp16 1.0 splat B-fragment
#pragma unroll
    for (int v = 0; v < 8; ++v) ones.u[v] = 0x3C003C00u;

    float mrow[8];
    v8f zero = {};
    v8f acc[4] = { zero, zero, zero, zero };
    v8f lacc = zero;                       // per-row softmax denominators (via WMMA)
#pragma unroll
    for (int v = 0; v < 8; ++v) mrow[v] = -3.0e38f;

    __shared__ __align__(16) u16 pb[16 * 64];

    for (int s0 = 0; s0 < Tx; s0 += 64) {
        // scores for 64 keys: 4 sim fragments, each K=d=64 via 2 chained WMMAs
        v8f sim[4];
#pragma unroll
        for (int j = 0; j < 4; ++j) {
            sim[j] = zero;
            const u16* kr = kt + (size_t)(s0 + j * 16 + m) * Cx;
#pragma unroll
            for (int f = 0; f < 2; ++f) {
                Frag bk;
                const u16* p = kr + f * 32 + (hl << 4);
                bk.q[0] = *(const u32x4*)(p);
                bk.q[1] = *(const u32x4*)(p + 8);
                sim[j] = __builtin_amdgcn_wmma_f32_16x16x32_f16(
                    false, aq[f].h, false, bk.h, (short)0, sim[j], false, false);
            }
        }

        // online max (only reduction left): one 4-step butterfly per row per 64 keys
        float corr[8];
#pragma unroll
        for (int v = 0; v < 8; ++v) {
            float tmax = fmaxf(fmaxf(sim[0][v], sim[1][v]), fmaxf(sim[2][v], sim[3][v]));
            tmax = fmaxf(tmax, __shfl_xor(tmax, 1, 32));
            tmax = fmaxf(tmax, __shfl_xor(tmax, 2, 32));
            tmax = fmaxf(tmax, __shfl_xor(tmax, 4, 32));
            tmax = fmaxf(tmax, __shfl_xor(tmax, 8, 32));
            float mn = fmaxf(mrow[v], tmax);
            corr[v] = __expf(mrow[v] - mn);
            mrow[v] = mn;
            int r = v + (hl << 3);
#pragma unroll
            for (int j = 0; j < 4; ++j)
                pb[r * 64 + j * 16 + m] = f2h(__expf(sim[j][v] - mn));
        }
#pragma unroll
        for (int v = 0; v < 8; ++v) {
            lacc[v] *= corr[v];
#pragma unroll
            for (int j = 0; j < 4; ++j) acc[j][v] *= corr[v];
        }

        asm volatile("s_wait_dscnt 0" ::: "memory");

        // P as two A-fragments (k=0..31, 32..63): 4 x ds_load_b128
        Frag ap[2];
#pragma unroll
        for (int f = 0; f < 2; ++f) {
            ap[f].q[0] = *(const u32x4*)(pb + m * 64 + f * 32 + (hl << 4));
            ap[f].q[1] = *(const u32x4*)(pb + m * 64 + f * 32 + 8 + (hl << 4));
        }

        // denominators: lacc += P * ones  (every lane gets its own row sum)
#pragma unroll
        for (int f = 0; f < 2; ++f)
            lacc = __builtin_amdgcn_wmma_f32_16x16x32_f16(
                false, ap[f].h, false, ones.h, (short)0, lacc, false, false);

        // acc[j] += P * V_j over K=64 (V k-dim contiguous -> 2 x b128 per fragment)
#pragma unroll
        for (int j = 0; j < 4; ++j) {
            const u16* vr = vb + (size_t)(j * 16 + m) * Tx + s0;
#pragma unroll
            for (int f = 0; f < 2; ++f) {
                Frag bv;
                const u16* p = vr + f * 32 + (hl << 4);
                bv.q[0] = *(const u32x4*)(p);
                bv.q[1] = *(const u32x4*)(p + 8);
                acc[j] = __builtin_amdgcn_wmma_f32_16x16x32_f16(
                    false, ap[f].h, false, bv.h, (short)0, acc[j], false, false);
            }
        }
    }

    // epilogue: divide by per-row denominators (already lane-local); store t-major
#pragma unroll
    for (int v = 0; v < 8; ++v) {
        float inv = 1.0f / lacc[v];
        int r = v + (hl << 3);
#pragma unroll
        for (int j = 0; j < 4; ++j)
            ot[(size_t)(t0 + r) * Cx + j * 16 + m] = f2h(acc[j][v] * inv);
    }
}

// ------------------------------------------------- BN stats (per channel over B,T)
__global__ __launch_bounds__(256) void bn_stats(const float* __restrict__ y,
                                                float* __restrict__ mean,
                                                float* __restrict__ rstd)
{
    const int c = blockIdx.x;
    __shared__ float s1[256], s2[256];
    float a = 0.f, q = 0.f;
    for (int idx = threadIdx.x; idx < Bx * Tx; idx += 256) {
        int bb = idx >> 10, t = idx & 1023;
        float v = y[((size_t)bb * Cx + c) * Tx + t];
        a += v; q += v * v;
    }
    s1[threadIdx.x] = a; s2[threadIdx.x] = q;
    __syncthreads();
    for (int st = 128; st > 0; st >>= 1) {
        if (threadIdx.x < st) {
            s1[threadIdx.x] += s1[threadIdx.x + st];
            s2[threadIdx.x] += s2[threadIdx.x + st];
        }
        __syncthreads();
    }
    if (threadIdx.x == 0) {
        float inv = 1.0f / (float)(Bx * Tx);
        float mu = s1[0] * inv;
        float var = s2[0] * inv - mu * mu;
        mean[c] = mu;
        rstd[c] = rsqrtf(var + EPSx);
    }
}

// ------------------------------------------------- BN apply + residual
__global__ __launch_bounds__(256) void bn_res(const float* __restrict__ x,
                                              const float* __restrict__ y,
                                              const float* __restrict__ mean,
                                              const float* __restrict__ rstd,
                                              const float* __restrict__ gamma,
                                              const float* __restrict__ beta,
                                              float* __restrict__ out)
{
    size_t i = (size_t)blockIdx.x * 256 + threadIdx.x;
    int c = (int)((i >> 10) & (Cx - 1));
    out[i] = x[i] + (y[i] - mean[c]) * rstd[c] * gamma[c] + beta[c];
}

// ------------------------------------------------- launch
extern "C" void kernel_launch(void* const* d_in, const int* in_sizes, int n_in,
                              void* d_out, int out_size, void* d_ws, size_t ws_size,
                              hipStream_t stream) {
    const float* x     = (const float*)d_in[0];
    const float* wq    = (const float*)d_in[1];
    const float* bq    = (const float*)d_in[2];
    const float* wk    = (const float*)d_in[3];
    const float* bk    = (const float*)d_in[4];
    const float* wv    = (const float*)d_in[5];
    const float* bv    = (const float*)d_in[6];
    const float* wo    = (const float*)d_in[7];
    const float* bo    = (const float*)d_in[8];
    const float* gamma = (const float*)d_in[9];
    const float* beta  = (const float*)d_in[10];
    float* out = (float*)d_out;

    const size_t MB = (size_t)1 << 20;
    char* w = (char*)d_ws;
    u16*   XT   = (u16*)(w);             // 16 MB  x^T (B,T,C) fp16
    u16*   WQH  = (u16*)(w + 16 * MB);   //  2 MB each
    u16*   WKH  = (u16*)(w + 18 * MB);
    u16*   WVH  = (u16*)(w + 20 * MB);
    u16*   WOH  = (u16*)(w + 22 * MB);
    u16*   QT   = (u16*)(w + 24 * MB);   // 16 MB (B,T,C), pre-scaled by SCALE
    u16*   KT   = (u16*)(w + 40 * MB);   // 16 MB (B,T,C)
    u16*   VH   = (u16*)(w + 56 * MB);   // 16 MB (B,C,T)
    u16*   OT   = (u16*)(w + 72 * MB);   // 16 MB attention out (B,T,C)
    float* Y    = (float*)(w + 88 * MB); // 32 MB pre-BN projection (B,C,T) fp32
    float* MEAN = (float*)(w + 120 * MB);
    float* RSTD = (float*)(w + 120 * MB + 4096);

    const int NX = Bx * Cx * Tx;   // 8M
    const int NW = Cx * Cx;        // 1M

    cvt_transpose<<<dim3(Tx / 32, Cx / 32, Bx), dim3(32, 8), 0, stream>>>(x, XT);
    cvt_f32_f16<<<(NW + 255) / 256, 256, 0, stream>>>(wq, WQH, NW);
    cvt_f32_f16<<<(NW + 255) / 256, 256, 0, stream>>>(wk, WKH, NW);
    cvt_f32_f16<<<(NW + 255) / 256, 256, 0, stream>>>(wv, WVH, NW);
    cvt_f32_f16<<<(NW + 255) / 256, 256, 0, stream>>>(wo, WOH, NW);

    dim3 ggrid(Tx / 64, Cx / 128, Bx);
    dim3 gblk(32, 4);
    gemm_proj<2><<<ggrid, gblk, 0, stream>>>(WQH, XT, bq, SCALEx, QT, nullptr);
    gemm_proj<2><<<ggrid, gblk, 0, stream>>>(WKH, XT, bk, 1.0f,   KT, nullptr);
    gemm_proj<1><<<ggrid, gblk, 0, stream>>>(WVH, XT, bv, 1.0f,   VH, nullptr);

    flash_attn<<<dim3(Tx / 16, Hx, Bx), 32, 0, stream>>>(QT, KT, VH, OT);

    gemm_proj<0><<<ggrid, gblk, 0, stream>>>(WOH, OT, bo, 1.0f, nullptr, Y);

    bn_stats<<<Cx, 256, 0, stream>>>(Y, MEAN, RSTD);
    bn_res<<<NX / 256, 256, 0, stream>>>(x, Y, MEAN, RSTD, gamma, beta, out);
}
